// Model_10668698764021
// MI455X (gfx1250) — matrix operand order repaired and checked by reading, not compiled, not run on hardware
//
#include <hip/hip_runtime.h>
#include <hip/hip_bf16.h>
#include <math.h>

typedef __attribute__((ext_vector_type(16))) __bf16 v16bf;
typedef __attribute__((ext_vector_type(8)))  __bf16 v8bf;
typedef __attribute__((ext_vector_type(8)))  float  v8f;
typedef __attribute__((ext_vector_type(4)))  unsigned int u32x4;
typedef __attribute__((ext_vector_type(8)))  int          i32x8;
typedef __attribute__((ext_vector_type(4)))  int          i32x4;

static constexpr int    SEQ     = 2048;
static constexpr int    DMODEL  = 1024;
static constexpr int    DINNER  = 2048;
static constexpr int    NH      = 32;
static constexpr int    HD      = 64;
static constexpr int    DSTATE  = 128;
static constexpr int    CONVDIM = 2304;
static constexpr int    DINPROJ = 4384;
static constexpr int    VOCABN  = 32000;
static constexpr int    NLAYER  = 4;

#if defined(__has_builtin)
#  if __has_builtin(__builtin_amdgcn_tensor_load_to_lds) && \
      __has_builtin(__builtin_amdgcn_s_wait_tensorcnt)
#    define HAVE_TDM 1
#  endif
#endif
#ifndef HAVE_TDM
#  define HAVE_TDM 0
#endif

// ---------------- helpers ----------------
__device__ __forceinline__ unsigned short f2bf(float f) {
  unsigned u = __builtin_bit_cast(unsigned, f);
  u += 0x7fffu + ((u >> 16) & 1u);               // round-to-nearest-even
  return (unsigned short)(u >> 16);
}

__device__ __forceinline__ float blockReduceSum(float v, float* red) {
  #pragma unroll
  for (int off = 16; off > 0; off >>= 1) v += __shfl_xor(v, off, 32);
  const int lane = threadIdx.x & 31, wave = threadIdx.x >> 5;
  if (lane == 0) red[wave] = v;
  __syncthreads();
  v = (lane < (int)(blockDim.x >> 5)) ? red[lane] : 0.f;
  #pragma unroll
  for (int off = 4; off > 0; off >>= 1) v += __shfl_xor(v, off, 32);
  return __shfl(v, 0, 32);
}

#if HAVE_TDM
// Issue a TDM load of 256 contiguous f32 (one [B_t|C_t] segment) into LDS.
// Descriptor packing per CDNA5 ISA 8.3/8.4:
//  group0: count=1, lds_addr, global_addr[56:0], type=2 ("image")
//  group1: wg_mask=0 (not in cluster), data_size=4B, tile_dim0=256,
//          tensor_dim0=256, tensor_dim0_stride=256, tensor_dim1=1
__device__ __forceinline__ void tdm_load_row256(const float* gsrc, float* ldsDst) {
  const unsigned long long ga = (unsigned long long)(size_t)gsrc;
  const unsigned lds = (unsigned)(size_t)ldsDst;   // low bits = LDS byte offset
  u32x4 g0 = { 1u,                                  // count=1, user descriptor
               lds,                                 // bits 63:32  lds_addr
               (unsigned)(ga & 0xffffffffu),        // bits 95:64  global_addr lo
               (unsigned)((ga >> 32) & 0x01ffffffu) // bits 120:96 global_addr hi
                   | (2u << 30) };                  // bits 127:126 type=2
  i32x8 g1 = { (int)(2u << 16),                     // data_size = 4 bytes
               (int)(256u << 16),                   // tensor_dim0[15:0] = 256
               (int)(1u << 16),                     // tensor_dim0 hi=0, tensor_dim1=1
               (int)(256u << 16),                   // tensor_dim1 hi=0, tile_dim0=256
               0,                                   // tile_dim1=0, tile_dim2=0
               256,                                 // tensor_dim0_stride = 256
               0, 0 };
  i32x4 gz = { 0, 0, 0, 0 };
#if __clang_major__ >= 23
  i32x8 gz8 = { 0, 0, 0, 0, 0, 0, 0, 0 };
  __builtin_amdgcn_tensor_load_to_lds(g0, g1, gz, gz, gz8, 0);
#else
  __builtin_amdgcn_tensor_load_to_lds(g0, g1, gz, gz, 0);
#endif
}
#endif

// ---------------- elementwise / small kernels ----------------
__global__ __launch_bounds__(256) void k_f32_to_bf16(
    const float* __restrict__ src, unsigned short* __restrict__ dst, int n) {
  for (int i = blockIdx.x * blockDim.x + threadIdx.x; i < n;
       i += gridDim.x * blockDim.x)
    dst[i] = f2bf(src[i]);
}

__global__ __launch_bounds__(256) void k_embed_norm(
    const int* __restrict__ idx, const float* __restrict__ emb,
    float* __restrict__ x) {
  __shared__ float red[8];
  const int t = blockIdx.x;
  const float* e = emb + (size_t)idx[t] * DMODEL;
  float v[4], ss = 0.f;
  #pragma unroll
  for (int j = 0; j < 4; ++j) {
    v[j] = e[threadIdx.x + j * 256];
    ss += v[j] * v[j];
  }
  ss = blockReduceSum(ss, red);
  const float inv = rsqrtf(ss * (1.f / DMODEL) + 1.1920929e-07f);
  #pragma unroll
  for (int j = 0; j < 4; ++j)
    x[(size_t)t * DMODEL + threadIdx.x + j * 256] = v[j] * inv;
}

__global__ __launch_bounds__(256) void k_rmsnorm_bf16(
    const float* __restrict__ x, const float* __restrict__ w,
    unsigned short* __restrict__ out) {
  __shared__ float red[8];
  const int t = blockIdx.x;
  float v[4], ss = 0.f;
  #pragma unroll
  for (int j = 0; j < 4; ++j) {
    v[j] = x[(size_t)t * DMODEL + threadIdx.x + j * 256];
    ss += v[j] * v[j];
  }
  ss = blockReduceSum(ss, red);
  const float inv = rsqrtf(ss * (1.f / DMODEL) + 1e-5f);
  #pragma unroll
  for (int j = 0; j < 4; ++j) {
    const int c = threadIdx.x + j * 256;
    out[(size_t)t * DMODEL + c] = f2bf(v[j] * inv * w[c]);
  }
}

__global__ __launch_bounds__(256) void k_conv_silu(
    const float* __restrict__ zx, const float* __restrict__ cw,
    const float* __restrict__ cb, float* __restrict__ xbc) {
  const int i = blockIdx.x * blockDim.x + threadIdx.x;
  if (i >= SEQ * CONVDIM) return;
  const int t = i / CONVDIM, c = i % CONVDIM;
  float acc = cb[c];
  #pragma unroll
  for (int j = 0; j < 4; ++j) {
    const int tt = t - 3 + j;
    if (tt >= 0) acc += zx[(size_t)tt * DINPROJ + DINNER + c] * cw[c * 4 + j];
  }
  xbc[i] = acc / (1.f + __expf(-acc));          // silu
}

__global__ __launch_bounds__(256) void k_dt(
    const float* __restrict__ zx, const float* __restrict__ dtb,
    const float* __restrict__ alog, float* __restrict__ dtv,
    float* __restrict__ dav) {
  const int i = blockIdx.x * blockDim.x + threadIdx.x;
  if (i >= SEQ * NH) return;
  const int hh = i & 31;
  const float xr = zx[(size_t)(i >> 5) * DINPROJ + (DINPROJ - NH) + hh] + dtb[hh];
  const float sp = (xr > 20.f) ? xr : log1pf(__expf(xr));   // softplus
  dtv[i] = sp;
  dav[i] = __expf(-__expf(alog[hh]) * sp);                  // exp(dt*A), A=-exp(A_log)
}

// sequential SSM scan: one block per head, state in registers.
// B_t/C_t (256 f32, contiguous in the conv-out row) are staged into a
// double-buffered LDS slot one step ahead via the Tensor Data Mover when
// available; wave 0 issues, s_wait_tensorcnt + barrier publishes.
__global__ __launch_bounds__(256) void k_scan(
    const float* __restrict__ xbc, const float* __restrict__ dtv,
    const float* __restrict__ dav, const float* __restrict__ Dp,
    float* __restrict__ y) {
  __shared__ float ldsBC[2][2 * DSTATE];
  const int head = blockIdx.x;
  const int tid = threadIdx.x;
  const int p = tid >> 2;               // 0..63
  const int ng = tid & 3;               // quarter of state dim
  const float Dh = Dp[head];
  float h[32];
  #pragma unroll
  for (int i = 0; i < 32; ++i) h[i] = 0.f;

#if HAVE_TDM
  if (tid < 32) {                       // wave 0 only (TDM ignores EXEC)
    tdm_load_row256(xbc + DINNER, &ldsBC[0][0]);
    __builtin_amdgcn_s_wait_tensorcnt(0);
  }
  __syncthreads();
#endif

  for (int t = 0; t < SEQ; ++t) {
    const float* row = xbc + (size_t)t * CONVDIM;
    const int cur = t & 1;
#if HAVE_TDM
    if (tid < 32 && t + 1 < SEQ)        // prefetch next step's [B|C] segment
      tdm_load_row256(xbc + (size_t)(t + 1) * CONVDIM + DINNER,
                      &ldsBC[cur ^ 1][0]);
#else
    __syncthreads();
    ldsBC[cur][tid] = row[DINNER + tid];   // 256 threads, 256 floats
    __syncthreads();
#endif
    const float dA = dav[t * NH + head];
    const float dt = dtv[t * NH + head];
    const float xv = row[head * HD + p];
    const float xdt = xv * dt;
    const float* Bv = &ldsBC[cur][0];
    const float* Cv = &ldsBC[cur][DSTATE];
    float acc = 0.f;
    const int nb = ng * 32;
    #pragma unroll
    for (int i = 0; i < 32; ++i) {
      h[i] = h[i] * dA + xdt * Bv[nb + i];
      acc += h[i] * Cv[nb + i];
    }
    acc += __shfl_xor(acc, 1, 32);      // 4 consecutive lanes share p
    acc += __shfl_xor(acc, 2, 32);
    if (ng == 0) y[(size_t)t * DINNER + head * HD + p] = acc + Dh * xv;
#if HAVE_TDM
    if (tid < 32) __builtin_amdgcn_s_wait_tensorcnt(0);
    __syncthreads();                    // next buffer visible to all waves
#endif
  }
}

__global__ __launch_bounds__(256) void k_gate_norm(
    const float* __restrict__ y, const float* __restrict__ zx,
    const float* __restrict__ gw, unsigned short* __restrict__ out) {
  __shared__ float g[DINNER];
  __shared__ float red[8];
  const int t = blockIdx.x;
  float ss = 0.f;
  for (int i = threadIdx.x; i < DINNER; i += 256) {
    const float z = zx[(size_t)t * DINPROJ + i];
    const float v = y[(size_t)t * DINNER + i] * (z / (1.f + __expf(-z)));
    g[i] = v;
    ss += v * v;
  }
  ss = blockReduceSum(ss, red);
  const float inv = rsqrtf(ss * (1.f / DINNER) + 1e-5f);
  for (int i = threadIdx.x; i < DINNER; i += 256)
    out[(size_t)t * DINNER + i] = f2bf(g[i] * inv * gw[i]);
}

__global__ __launch_bounds__(256) void k_add(
    float* __restrict__ x, const float* __restrict__ h, int n) {
  for (int i = blockIdx.x * blockDim.x + threadIdx.x; i < n;
       i += gridDim.x * blockDim.x)
    x[i] += h[i];
}

// ---------------- bf16 WMMA GEMM: C[M,N] f32 = A[M,K] * W[N,K]^T ----------------
// block = 256 threads = 8 waves stacked along M; wave tile 32 x (NFRAG*16);
// block tile 256 x 64. NFRAG is a compile-time constant so the inner loop is
// branch-free and fully unrolled (no accumulator shadow copies).
// Fragment layouts per CDNA5 ISA 7.12.2 (16-bit A 16x32, B 32x16, f32 C/D 16x16).
template <int NFRAG>
__global__ __launch_bounds__(256) void k_gemm_bf16(
    const unsigned short* __restrict__ A, const unsigned short* __restrict__ W,
    float* __restrict__ C, int M, int N, int K, int n0Base) {
  const int lane   = threadIdx.x & 31;
  const int wave   = threadIdx.x >> 5;
  const int laneLo = lane & 15;
  const int laneHi = lane >> 4;
  const int m0 = blockIdx.y * 256 + wave * 32;
  const int n0 = n0Base + blockIdx.x * 64;

  v8f acc[2][NFRAG] = {};
  const unsigned short* aRow0 = A + (size_t)(m0 + laneLo) * K + laneHi * 8;
  const unsigned short* aRow1 = aRow0 + (size_t)16 * K;
  const unsigned short* wRow[NFRAG];
  #pragma unroll
  for (int j = 0; j < NFRAG; ++j)
    wRow[j] = W + (size_t)(n0 + j * 16 + laneLo) * K + laneHi * 8;

  for (int k0 = 0; k0 < K; k0 += 32) {
    // A fragment: lane<16 -> K {0..7,16..23}; lane>=16 -> K {8..15,24..31}
    v8bf a0lo = *(const v8bf*)(aRow0 + k0);
    v8bf a0hi = *(const v8bf*)(aRow0 + k0 + 16);
    v8bf a1lo = *(const v8bf*)(aRow1 + k0);
    v8bf a1hi = *(const v8bf*)(aRow1 + k0 + 16);
    v16bf a0 = __builtin_shufflevector(a0lo, a0hi,
        0,1,2,3,4,5,6,7,8,9,10,11,12,13,14,15);
    v16bf a1 = __builtin_shufflevector(a1lo, a1hi,
        0,1,2,3,4,5,6,7,8,9,10,11,12,13,14,15);
    __builtin_prefetch(aRow0 + k0 + 256, 0, 1);   // global_prefetch_b8
    __builtin_prefetch(aRow1 + k0 + 256, 0, 1);
    #pragma unroll
    for (int j = 0; j < NFRAG; ++j) {
      v8bf blo = *(const v8bf*)(wRow[j] + k0);
      v8bf bhi = *(const v8bf*)(wRow[j] + k0 + 16);
      v16bf b = __builtin_shufflevector(blo, bhi,
          0,1,2,3,4,5,6,7,8,9,10,11,12,13,14,15);
      __builtin_prefetch(wRow[j] + k0 + 256, 0, 1);
      acc[0][j] = __builtin_amdgcn_wmma_f32_16x16x32_bf16(
          false, a0, false, b, (short)0, acc[0][j], false, false);
      acc[1][j] = __builtin_amdgcn_wmma_f32_16x16x32_bf16(
          false, a1, false, b, (short)0, acc[1][j], false, false);
    }
  }
  #pragma unroll
  for (int mi = 0; mi < 2; ++mi) {
    #pragma unroll
    for (int j = 0; j < NFRAG; ++j) {
      float* cp = C + (size_t)(m0 + mi * 16 + laneHi * 8) * N
                    + (n0 + j * 16 + laneLo);
      #pragma unroll
      for (int r = 0; r < 8; ++r) cp[(size_t)r * N] = acc[mi][j][r];
    }
  }
}

// ---------------- workspace layout (bytes) ----------------
static constexpr size_t OFF_WBF = 0;                                   // bf16 weights (max: lm_head)
static constexpr size_t SZ_WBF  = (size_t)VOCABN * DMODEL * 2;         // 65.5 MB
static constexpr size_t OFF_X   = OFF_WBF + SZ_WBF;                    // residual stream f32
static constexpr size_t OFF_XN  = OFF_X   + (size_t)SEQ * DMODEL * 4;  // normed acts bf16
static constexpr size_t OFF_Z   = OFF_XN  + (size_t)SEQ * DMODEL * 2;  // zxbcdt f32
static constexpr size_t OFF_XBC = OFF_Z   + (size_t)SEQ * DINPROJ * 4; // conv out f32
static constexpr size_t OFF_DT  = OFF_XBC + (size_t)SEQ * CONVDIM * 4;
static constexpr size_t OFF_DA  = OFF_DT  + (size_t)SEQ * NH * 4;
static constexpr size_t OFF_Y   = OFF_DA  + (size_t)SEQ * NH * 4;      // scan out f32
static constexpr size_t OFF_YG  = OFF_Y   + (size_t)SEQ * DINNER * 4;  // gated bf16
static constexpr size_t OFF_TMP = OFF_YG  + (size_t)SEQ * DINNER * 2;  // out_proj f32

extern "C" void kernel_launch(void* const* d_in, const int* in_sizes, int n_in,
                              void* d_out, int out_size, void* d_ws, size_t ws_size,
                              hipStream_t stream) {
  const int*   idx    = (const int*)  d_in[0];
  const float* emb    = (const float*)d_in[1];
  const float* norm_w = (const float*)d_in[2];   // (4,1024)
  const float* in_w   = (const float*)d_in[3];   // (4,4384,1024)
  const float* conv_w = (const float*)d_in[4];   // (4,2304,4)
  const float* conv_b = (const float*)d_in[5];   // (4,2304)
  const float* dt_b   = (const float*)d_in[6];   // (4,32)
  const float* a_log  = (const float*)d_in[7];   // (4,32)
  const float* Dpar   = (const float*)d_in[8];   // (4,32)
  const float* gnw    = (const float*)d_in[9];   // (4,2048)
  const float* out_w  = (const float*)d_in[10];  // (4,1024,2048)
  const float* nfw    = (const float*)d_in[11];  // (1024,)
  const float* lmw    = (const float*)d_in[12];  // (32000,1024)

  char* ws = (char*)d_ws;
  unsigned short* WBF = (unsigned short*)(ws + OFF_WBF);
  float*          X   = (float*)         (ws + OFF_X);
  unsigned short* XN  = (unsigned short*)(ws + OFF_XN);
  float*          Z   = (float*)         (ws + OFF_Z);
  float*          XBC = (float*)         (ws + OFF_XBC);
  float*          DT  = (float*)         (ws + OFF_DT);
  float*          DA  = (float*)         (ws + OFF_DA);
  float*          Y   = (float*)         (ws + OFF_Y);
  unsigned short* YG  = (unsigned short*)(ws + OFF_YG);
  float*          TMP = (float*)         (ws + OFF_TMP);

  k_embed_norm<<<SEQ, 256, 0, stream>>>(idx, emb, X);

  for (int l = 0; l < NLAYER; ++l) {
    k_rmsnorm_bf16<<<SEQ, 256, 0, stream>>>(X, norm_w + l * DMODEL, XN);

    const int nInW = DINPROJ * DMODEL;
    k_f32_to_bf16<<<2048, 256, 0, stream>>>(in_w + (size_t)l * nInW, WBF, nInW);
    // 4384 = 68*64 + 32 -> full tiles with NFRAG=4, edge 32 cols with NFRAG=2
    k_gemm_bf16<4><<<dim3(68, SEQ / 256), 256, 0, stream>>>(
        XN, WBF, Z, SEQ, DINPROJ, DMODEL, 0);
    k_gemm_bf16<2><<<dim3(1, SEQ / 256), 256, 0, stream>>>(
        XN, WBF, Z, SEQ, DINPROJ, DMODEL, 68 * 64);

    k_conv_silu<<<(SEQ * CONVDIM + 255) / 256, 256, 0, stream>>>(
        Z, conv_w + (size_t)l * CONVDIM * 4, conv_b + (size_t)l * CONVDIM, XBC);
    k_dt<<<(SEQ * NH + 255) / 256, 256, 0, stream>>>(
        Z, dt_b + l * NH, a_log + l * NH, DT, DA);
    k_scan<<<NH, 256, 0, stream>>>(XBC, DT, DA, Dpar + l * NH, Y);
    k_gate_norm<<<SEQ, 256, 0, stream>>>(Y, Z, gnw + (size_t)l * DINNER, YG);

    const int nOutW = DMODEL * DINNER;
    k_f32_to_bf16<<<2048, 256, 0, stream>>>(out_w + (size_t)l * nOutW, WBF, nOutW);
    k_gemm_bf16<4><<<dim3(DMODEL / 64, SEQ / 256), 256, 0, stream>>>(
        YG, WBF, TMP, SEQ, DMODEL, DINNER, 0);
    k_add<<<2048, 256, 0, stream>>>(X, TMP, SEQ * DMODEL);
  }

  k_rmsnorm_bf16<<<SEQ, 256, 0, stream>>>(X, nfw, XN);
  const int nLmW = VOCABN * DMODEL;
  k_f32_to_bf16<<<4096, 256, 0, stream>>>(lmw, WBF, nLmW);
  k_gemm_bf16<4><<<dim3(VOCABN / 64, SEQ / 256), 256, 0, stream>>>(
      XN, WBF, (float*)d_out, SEQ, VOCABN, DMODEL, 0);
}
